// AffineCoupling_72035191488687
// MI455X (gfx1250) — matrix operand order repaired and checked by reading, not compiled, not run on hardware
//
#include <hip/hip_runtime.h>
#include <hip/hip_bf16.h>

typedef __bf16 bf16_t;
typedef __attribute__((ext_vector_type(16))) __bf16 v16bf;
typedef __attribute__((ext_vector_type(8)))  __bf16 v8bf;
typedef __attribute__((ext_vector_type(8)))  float  v8f;
typedef __attribute__((ext_vector_type(4)))  int    v4i;

#define BM 128
#define BN 128
#define BK 32
#define LDA_S 40   // padded LDS row stride (bf16 elems): 80B/row, keeps 16B alignment
#define LDB_S 40

#if __has_builtin(__builtin_amdgcn_global_load_async_to_lds_b128) && \
    __has_builtin(__builtin_amdgcn_s_wait_asynccnt)
#define USE_ASYNC 1
#else
#define USE_ASYNC 0
#endif

#if USE_ASYNC
// global -> LDS async copy of one b128, bypassing VGPRs (ASYNCcnt-tracked)
__device__ __forceinline__ void gld_async_b128(const void* g, void* l) {
  __builtin_amdgcn_global_load_async_to_lds_b128(
      (__attribute__((address_space(1))) v4i*)(void*)(g),
      (__attribute__((address_space(3))) v4i*)(l),
      0, 0);
}
__device__ __forceinline__ void tile_async(const bf16_t* gA, const bf16_t* gB,
                                           bf16_t* sA, bf16_t* sB) {
  gld_async_b128(gA,     sA);
  gld_async_b128(gA + 8, sA + 8);
  gld_async_b128(gB,     sB);
  gld_async_b128(gB + 8, sB + 8);
}
#endif

// ---------- prep: x1 = bf16(x[:, 0::2]) ----------
__global__ void k_pack_x1(const float* __restrict__ x, bf16_t* __restrict__ x1,
                          int B, int W) {
  int idx = blockIdx.x * blockDim.x + threadIdx.x;
  int half = W >> 1;
  if (idx < B * half) {
    int b = idx / half, j = idx % half;
    x1[idx] = (bf16_t)x[(size_t)b * W + 2 * j];
  }
}

// ---------- prep: wT[n*K + k] = bf16(w[k*N + n]) ----------
__global__ void k_wT(const float* __restrict__ w, bf16_t* __restrict__ wT,
                     int K, int N) {
  int idx = blockIdx.x * blockDim.x + threadIdx.x;
  if (idx < K * N) {
    int n = idx / K, k = idx % K;
    wT[idx] = (bf16_t)w[(size_t)k * N + n];
  }
}

// fragment loads + 8 WMMAs from LDS buffer `curbuf` (macro so both call
// sites share identical register allocation for the accumulators)
#define GEMM_STEP(curbuf)                                                     \
  do {                                                                        \
    const bf16_t* AsC = As[(curbuf)];                                         \
    const bf16_t* BsC = Bs[(curbuf)];                                         \
    /* A frag 16x32: lanes 0-15 K{0..7,16..23}; lanes 16-31 K{8..15,24..31} */\
    V16 afrag[2];                                                             \
    _Pragma("unroll")                                                         \
    for (int mt = 0; mt < 2; mt++) {                                          \
      const bf16_t* s = &AsC[(wm * 32 + mt * 16 + lrow) * LDA_S];             \
      afrag[mt].h[0] = *(const v8bf*)(s + hi * 8);                            \
      afrag[mt].h[1] = *(const v8bf*)(s + 16 + hi * 8);                       \
    }                                                                         \
    /* B frag 32x16: lanes 0-15 K{0..7,16..23}; lanes 16-31 K{8..15,24..31} */\
    V16 bfrag[4];                                                             \
    _Pragma("unroll")                                                         \
    for (int nt = 0; nt < 4; nt++) {                                          \
      const bf16_t* s = &BsC[(wn * 64 + nt * 16 + lrow) * LDB_S];             \
      bfrag[nt].h[0] = *(const v8bf*)(s + hi * 8);                            \
      bfrag[nt].h[1] = *(const v8bf*)(s + 16 + hi * 8);                       \
    }                                                                         \
    _Pragma("unroll")                                                         \
    for (int mt = 0; mt < 2; mt++)                                            \
      _Pragma("unroll")                                                       \
      for (int nt = 0; nt < 4; nt++)                                          \
        acc[mt][nt] = __builtin_amdgcn_wmma_f32_16x16x32_bf16(                \
            false, afrag[mt].v, false, bfrag[nt].v,                           \
            (short)0, acc[mt][nt], false, false);                             \
  } while (0)

// ---------- WMMA GEMM: C = A(MxK,bf16) * Bt(NxK,bf16)^T + bias ----------
// relu!=0: out_bf16 = relu(C) (bf16).  relu==0: out_f32 = C (f32).
__global__ void __launch_bounds__(256)
k_gemm_wmma(const bf16_t* __restrict__ A, const bf16_t* __restrict__ Bt,
            const float* __restrict__ bias,
            bf16_t* __restrict__ out_bf16, float* __restrict__ out_f32,
            int M, int N, int K, int relu) {
  __shared__ bf16_t As[2][BM * LDA_S];
  __shared__ bf16_t Bs[2][BN * LDB_S];

  const int tid  = threadIdx.x;
  const int lane = tid & 31;
  const int wave = tid >> 5;
  const int wm   = wave & 3;   // 4 wave-rows  -> 32 M-rows each
  const int wn   = wave >> 2;  // 2 wave-cols  -> 64 N-cols each
  const int lrow = lane & 15;
  const int hi   = lane >> 4;  // lane group 0/1

  const int block_m = blockIdx.y * BM;
  const int block_n = blockIdx.x * BN;

  v8f acc[2][4];
  v8f zero = {};
#pragma unroll
  for (int i = 0; i < 2; i++)
#pragma unroll
    for (int j = 0; j < 4; j++) acc[i][j] = zero;

  union V16 { v16bf v; v8bf h[2]; };

  // per-thread slice of the cooperative tile copy: 32B of A and 32B of B
  const int row = tid >> 1;
  const int col = (tid & 1) * 16;
  const bf16_t* gA = A  + (size_t)(block_m + row) * K + col;
  const bf16_t* gB = Bt + (size_t)(block_n + row) * K + col;
  bf16_t* sA[2] = { &As[0][row * LDA_S + col], &As[1][row * LDA_S + col] };
  bf16_t* sB[2] = { &Bs[0][row * LDB_S + col], &Bs[1][row * LDB_S + col] };

  const int nIter = K / BK;

#if USE_ASYNC
  // prologue: tile 0 -> buffer 0
  tile_async(gA, gB, sA[0], sB[0]);
  int i = 0;
  // steady state: branch-free (issue next tile, wait for current, compute)
  for (; i + 1 < nIter; ++i) {
    const int cur = i & 1;
    tile_async(gA + (size_t)(i + 1) * BK, gB + (size_t)(i + 1) * BK,
               sA[cur ^ 1], sB[cur ^ 1]);
    __builtin_amdgcn_s_wait_asynccnt(4);   // tile i landed; tile i+1 in flight
    __syncthreads();
    GEMM_STEP(cur);
    __syncthreads();   // all waves done with buf[cur] before it is refilled
  }
  // peeled last iteration: drain everything, no further issue
  __builtin_amdgcn_s_wait_asynccnt(0);
  __syncthreads();
  GEMM_STEP(i & 1);
#else
  for (int i = 0; i < nIter; ++i) {
    const int cur = i & 1;
    {
      const bf16_t* g = gA + (size_t)i * BK;
      v8bf a0 = *(const v8bf*)(g);
      v8bf a1 = *(const v8bf*)(g + 8);
      *(v8bf*)(sA[cur])     = a0;
      *(v8bf*)(sA[cur] + 8) = a1;
      g = gB + (size_t)i * BK;
      v8bf b0 = *(const v8bf*)(g);
      v8bf b1 = *(const v8bf*)(g + 8);
      *(v8bf*)(sB[cur])     = b0;
      *(v8bf*)(sB[cur] + 8) = b1;
    }
    __syncthreads();
    GEMM_STEP(cur);
    __syncthreads();
  }
#endif

  // ---- epilogue: C/D layout: VGPR r, lanes 0-15 -> M=r, lanes 16-31 -> M=r+8; N=lane%16
#pragma unroll
  for (int mt = 0; mt < 2; mt++) {
#pragma unroll
    for (int nt = 0; nt < 4; nt++) {
      int mbase = block_m + wm * 32 + mt * 16 + hi * 8;
      int colO  = block_n + wn * 64 + nt * 16 + lrow;
      float bv  = bias[colO];
      if (relu) {
#pragma unroll
        for (int r = 0; r < 8; r++) {
          float v = acc[mt][nt][r] + bv;
          v = v > 0.0f ? v : 0.0f;
          out_bf16[(size_t)(mbase + r) * N + colO] = (bf16_t)v;
        }
      } else {
#pragma unroll
        for (int r = 0; r < 8; r++) {
          out_f32[(size_t)(mbase + r) * N + colO] = acc[mt][nt][r] + bv;
        }
      }
    }
  }
}

// ---------- final: y, log_det ----------
__global__ void __launch_bounds__(256)
k_final(const float* __restrict__ x, const float* __restrict__ ldJ,
        const float* __restrict__ bvec, float* __restrict__ y,
        float* __restrict__ log_det, int W) {
  int b = blockIdx.x;
  int half = W >> 1;
  float sum = 0.0f;
  for (int j = threadIdx.x; j < half; j += blockDim.x) {
    size_t base = (size_t)b * W + 2 * j;
    float b1 = tanhf(bvec[base]);
    float b2 = bvec[base + 1];
    float x1 = x[base];
    float x2 = x[base + 1];
    y[base]     = x1;
    y[base + 1] = x2 * __expf(b1) + b2;
    sum += b1;
  }
  __shared__ float red[256];
  red[threadIdx.x] = sum;
  __syncthreads();
  for (int s = 128; s > 0; s >>= 1) {
    if (threadIdx.x < s) red[threadIdx.x] += red[threadIdx.x + s];
    __syncthreads();
  }
  if (threadIdx.x == 0) log_det[b] = ldJ[b] + red[0];
}

extern "C" void kernel_launch(void* const* d_in, const int* in_sizes, int n_in,
                              void* d_out, int out_size, void* d_ws, size_t ws_size,
                              hipStream_t stream) {
  const float* x   = (const float*)d_in[0];
  const float* ldJ = (const float*)d_in[1];
  const float* w[6];
  const float* bias[6];
  for (int i = 0; i < 6; i++) {
    w[i]    = (const float*)d_in[2 + 2 * i];
    bias[i] = (const float*)d_in[3 + 2 * i];
  }

  const int B = 8192, W = 1024, MID = 2048, IN = W / 2;
  const int din[6]  = {IN, MID, MID, MID, MID, MID};
  const int dout[6] = {MID, MID, MID, MID, MID, W};

  // ---- workspace carve-up ----
  char* ws = (char*)d_ws;
  size_t off = 0;
  bf16_t* x1 = (bf16_t*)(ws + off); off += (size_t)B * IN * sizeof(bf16_t);
  bf16_t* wT[6];
  for (int i = 0; i < 6; i++) {
    wT[i] = (bf16_t*)(ws + off);
    off += (size_t)din[i] * dout[i] * sizeof(bf16_t);
  }
  bf16_t* actA = (bf16_t*)(ws + off); off += (size_t)B * MID * sizeof(bf16_t);
  bf16_t* actB = (bf16_t*)(ws + off); off += (size_t)B * MID * sizeof(bf16_t);
  float*  bvec = (float*)actB;  // final layer f32 output aliases the free ping buffer

  // ---- prep ----
  {
    int total = B * IN;
    k_pack_x1<<<(total + 255) / 256, 256, 0, stream>>>(x, x1, B, W);
  }
  for (int i = 0; i < 6; i++) {
    int total = din[i] * dout[i];
    k_wT<<<(total + 255) / 256, 256, 0, stream>>>(w[i], wT[i], din[i], dout[i]);
  }

  // ---- MLP chain ----
  const bf16_t* cur = x1;
  bf16_t* nxt = actA;
  for (int i = 0; i < 5; i++) {
    dim3 grid(dout[i] / BN, B / BM);
    k_gemm_wmma<<<grid, 256, 0, stream>>>(cur, wT[i], bias[i],
                                          nxt, (float*)nullptr,
                                          B, dout[i], din[i], 1);
    cur = nxt;
    nxt = (nxt == actA) ? actB : actA;
  }
  {
    dim3 grid(dout[5] / BN, B / BM);
    k_gemm_wmma<<<grid, 256, 0, stream>>>(cur, wT[5], bias[5],
                                          (bf16_t*)nullptr, bvec,
                                          B, dout[5], din[5], 0);
  }

  // ---- finalize: y + log_det ----
  float* y = (float*)d_out;
  float* log_det = y + (size_t)B * W;
  k_final<<<B, 256, 0, stream>>>(x, ldJ, bvec, y, log_det, W);
}
